// GRU_43576738185642
// MI455X (gfx1250) — matrix operand order repaired and checked by reading, not compiled, not run on hardware
//
#include <hip/hip_runtime.h>
#include <hip/hip_bf16.h>

// ---------------------------------------------------------------------------
// GRU recurrent net on gfx1250 (CDNA5), bf16 WMMA with f32 accumulate.
// d_in order: x, init_state, W_enc, W_wz, W_uz, W_wr, W_ur, W_wh, W_uh, W_dec
// d_out: h_1t (128*512*512 f32) ++ y_1t (128*512*512 f32)
// ---------------------------------------------------------------------------

typedef __attribute__((ext_vector_type(16))) __bf16 v16bf;
typedef __attribute__((ext_vector_type(8)))  float  v8f;

#define TT      512
#define NREC    512
#define BATCH   128
#define BBLK    16          // batch rows per workgroup == WMMA M
#define KTILES  16          // 512 / 32
#define NTHREADS 512        // 16 waves
#define NTW     2           // N-tiles (16 wide) per wave: 16*2*16 = 512
#define DT_OVER_TAU 0.2f

__device__ __forceinline__ unsigned short f2bf(float f) {
  unsigned u = __float_as_uint(f);
  u += 0x7FFFu + ((u >> 16) & 1u);          // round-to-nearest-even
  return (unsigned short)(u >> 16);
}
__device__ __forceinline__ float sigmoidf_(float v) { return 1.0f / (1.0f + __expf(-v)); }
__device__ __forceinline__ float tanhf_(float v)    { return 1.0f - 2.0f / (__expf(2.0f * v) + 1.0f); }

// WMMA A-layout (16-bit A 16x32, ISA 7.12.2), ktile-major LDS address in halves
__device__ __forceinline__ int a_addr(int m, int k) {
  int ktile = k >> 5, kk = k & 31;
  int hi   = (kk >> 3) & 1;                 // lanes 16-31 hold K 8-15 / 24-31
  int lane = m + (hi << 4);
  int e    = ((kk >> 4) << 3) + (kk & 7);   // halves 8-15 hold K 16-31
  return ((ktile << 5) + lane) * 16 + e;
}

__device__ __forceinline__ v8f wmma_bf16(v16bf a, v16bf b, v8f c) {
  return __builtin_amdgcn_wmma_f32_16x16x32_bf16(false, a, false, b, (short)0, c, false, false);
}

// ---------------------------------------------------------------------------
// Kernel 1: convert a 512x512 f32 weight (row-major, W[n][k], used as h@W^T)
// into bf16 WMMA B-layout fragments: frag(ntile,ktile)[lane][e] with
// k = ktile*32 + 8*(lane>>4) + (e&7) + 16*(e>>3) , n = ntile*16 + (lane&15)
// ---------------------------------------------------------------------------
__global__ __launch_bounds__(256) void swizzle_weight_bf16(
    const float* __restrict__ W, unsigned short* __restrict__ dst) {
  int f = blockIdx.x * 256 + threadIdx.x;           // 0 .. 262143
  if (f >= NREC * NREC) return;
  int fragIdx = f >> 9;                             // 512 halves per fragment
  int within  = f & 511;
  int lane    = within >> 4;
  int e       = within & 15;
  int ntile   = fragIdx >> 4;                       // fragIdx = ntile*16 + ktile
  int ktile   = fragIdx & 15;
  int k = ktile * 32 + ((lane >> 4) << 3) + (e & 7) + ((e >> 3) << 4);
  int n = ntile * 16 + (lane & 15);
  dst[f] = f2bf(W[n * NREC + k]);
}

// ---------------------------------------------------------------------------
// Kernel 2: persistent recurrence. One workgroup owns 16 batch rows for all
// 512 timesteps. h lives in LDS (fp32 + bf16 A-layout). Weights stream from
// L2 as pre-swizzled bf16 B fragments.
// ---------------------------------------------------------------------------
__global__ __launch_bounds__(NTHREADS, 1) void gru_recurrence(
    const float* __restrict__ x,            // (B, T, 2)
    const float* __restrict__ init_state,   // (B, 512)
    const float* __restrict__ Wwz,          // (512, 2)
    const float* __restrict__ Wwr,
    const float* __restrict__ Wwh,
    const unsigned short* __restrict__ Benc,  // swizzled bf16 weights
    const unsigned short* __restrict__ Buz,
    const unsigned short* __restrict__ Bur,
    const unsigned short* __restrict__ Buh,
    float* __restrict__ h_out)              // (B, T, 512)
{
  __shared__ __align__(32) unsigned short lds_a[KTILES * 32 * 16]; // 16 KB bf16 A-layout
  __shared__ float lds_h[BBLK][NREC];                              // 32 KB fp32 h
  __shared__ float lds_wz[NREC * 2];                               // 4 KB each
  __shared__ float lds_wr[NREC * 2];
  __shared__ float lds_wh[NREC * 2];
  __shared__ float lds_x[BBLK * 2];

  const int tid   = threadIdx.x;
  const int lane  = tid & 31;
  const int wave  = tid >> 5;               // 0..15
  const int bBase = blockIdx.x * BBLK;
  const int nl    = lane & 15;              // C/D: N within tile
  const int hiL   = lane >> 4;              // C/D: M = vgpr + 8*hiL

  // small input-projection weights -> LDS (once)
  for (int i = tid; i < NREC * 2; i += NTHREADS) {
    lds_wz[i] = Wwz[i]; lds_wr[i] = Wwr[i]; lds_wh[i] = Wwh[i];
  }
  // init_state block -> bf16 A-layout (A for the h0 = enc GEMM)
  for (int i = tid; i < BBLK * NREC; i += NTHREADS) {
    int m = i >> 9, k = i & 511;
    lds_a[a_addr(m, k)] = f2bf(init_state[(bBase + m) * NREC + k]);
  }
  __syncthreads();

  const v8f vzero = {0.f, 0.f, 0.f, 0.f, 0.f, 0.f, 0.f, 0.f};

  // ---- h0 = init_state @ W_enc^T  (K-split: 4 independent WMMA chains) ----
  {
    v8f acc0[NTW], acc1[NTW];
#pragma unroll
    for (int nt = 0; nt < NTW; ++nt) { acc0[nt] = vzero; acc1[nt] = vzero; }
#pragma unroll 4
    for (int kt = 0; kt < KTILES; kt += 2) {
      v16bf a0 = *(const v16bf*)&lds_a[(((kt + 0) << 5) + lane) * 16];
      v16bf a1 = *(const v16bf*)&lds_a[(((kt + 1) << 5) + lane) * 16];
#pragma unroll
      for (int nt = 0; nt < NTW; ++nt) {
        int ntile = wave * NTW + nt;
        v16bf b0 = *(const v16bf*)&Benc[(((ntile << 4) + kt + 0) << 9) + (lane << 4)];
        acc0[nt] = wmma_bf16(a0, b0, acc0[nt]);
        v16bf b1 = *(const v16bf*)&Benc[(((ntile << 4) + kt + 1) << 9) + (lane << 4)];
        acc1[nt] = wmma_bf16(a1, b1, acc1[nt]);
      }
    }
    __syncthreads();                       // all A reads done before overwrite
#pragma unroll
    for (int nt = 0; nt < NTW; ++nt) {
      int n = (wave * NTW + nt) * 16 + nl;
#pragma unroll
      for (int i = 0; i < 8; ++i) {
        int m = i + 8 * hiL;
        float h = acc0[nt][i] + acc1[nt][i];
        lds_h[m][n]         = h;
        lds_a[a_addr(m, n)] = f2bf(h);
      }
    }
    __syncthreads();
  }

  // ---- time loop ----
  for (int t = 0; t < TT; ++t) {
    if (tid < 32) {                        // x[b, t, 0..1] for this block
      int m = tid >> 1, c = tid & 1;
      lds_x[tid] = x[((size_t)(bBase + m) * TT + t) * 2 + c];
    }

    // --- Z and R GEMMs: A = h(bf16), B = W_uz / W_ur (4 chains) ---
    v8f zacc[NTW], racc[NTW];
#pragma unroll
    for (int nt = 0; nt < NTW; ++nt) { zacc[nt] = vzero; racc[nt] = vzero; }
#pragma unroll 2
    for (int kt = 0; kt < KTILES; ++kt) {
      v16bf a = *(const v16bf*)&lds_a[((kt << 5) + lane) * 16];
#pragma unroll
      for (int nt = 0; nt < NTW; ++nt) {
        int fb = (((wave * NTW + nt) << 4) + kt) << 9;
        v16bf bz = *(const v16bf*)&Buz[fb + (lane << 4)];
        zacc[nt] = wmma_bf16(a, bz, zacc[nt]);
        v16bf br = *(const v16bf*)&Bur[fb + (lane << 4)];
        racc[nt] = wmma_bf16(a, br, racc[nt]);
      }
    }
    __syncthreads();                       // h reads done; lds_x visible

    // --- gates z, r; write rh into A-layout for the U GEMM ---
    float zsave[NTW * 8];
#pragma unroll
    for (int nt = 0; nt < NTW; ++nt) {
      int n = (wave * NTW + nt) * 16 + nl;
      float wz0 = lds_wz[n * 2], wz1 = lds_wz[n * 2 + 1];
      float wr0 = lds_wr[n * 2], wr1 = lds_wr[n * 2 + 1];
#pragma unroll
      for (int i = 0; i < 8; ++i) {
        int m = i + 8 * hiL;
        float x0 = lds_x[m * 2], x1 = lds_x[m * 2 + 1];
        float hv = lds_h[m][n];
        float z  = sigmoidf_(x0 * wz0 + x1 * wz1 + zacc[nt][i]);
        float r  = sigmoidf_(x0 * wr0 + x1 * wr1 + racc[nt][i]);
        zsave[nt * 8 + i] = z;
        lds_a[a_addr(m, n)] = f2bf(r * hv);
      }
    }
    __syncthreads();                       // rh fully written

    // --- U GEMM: A = r*h, B = W_uh (K-split: 4 chains) ---
    v8f uacc0[NTW], uacc1[NTW];
#pragma unroll
    for (int nt = 0; nt < NTW; ++nt) { uacc0[nt] = vzero; uacc1[nt] = vzero; }
#pragma unroll 2
    for (int kt = 0; kt < KTILES; kt += 2) {
      v16bf a0 = *(const v16bf*)&lds_a[(((kt + 0) << 5) + lane) * 16];
      v16bf a1 = *(const v16bf*)&lds_a[(((kt + 1) << 5) + lane) * 16];
#pragma unroll
      for (int nt = 0; nt < NTW; ++nt) {
        int ntile = wave * NTW + nt;
        v16bf b0 = *(const v16bf*)&Buh[(((ntile << 4) + kt + 0) << 9) + (lane << 4)];
        uacc0[nt] = wmma_bf16(a0, b0, uacc0[nt]);
        v16bf b1 = *(const v16bf*)&Buh[(((ntile << 4) + kt + 1) << 9) + (lane << 4)];
        uacc1[nt] = wmma_bf16(a1, b1, uacc1[nt]);
      }
    }
    __syncthreads();                       // rh reads done before overwrite

    // --- leaky integration + outputs ---
#pragma unroll
    for (int nt = 0; nt < NTW; ++nt) {
      int n = (wave * NTW + nt) * 16 + nl;
      float wh0 = lds_wh[n * 2], wh1 = lds_wh[n * 2 + 1];
#pragma unroll
      for (int i = 0; i < 8; ++i) {
        int m = i + 8 * hiL;
        float x0 = lds_x[m * 2], x1 = lds_x[m * 2 + 1];
        float hv = lds_h[m][n];
        float u  = x0 * wh0 + x1 * wh1 + uacc0[nt][i] + uacc1[nt][i];
        float z  = zsave[nt * 8 + i];
        float hn = hv + DT_OVER_TAU * (z - 1.0f) * (hv - tanhf_(u));
        lds_h[m][n]         = hn;
        lds_a[a_addr(m, n)] = f2bf(hn);
        h_out[((size_t)(bBase + m) * TT + t) * NREC + n] = hn;
      }
    }
    __syncthreads();
  }
}

// ---------------------------------------------------------------------------
// Kernel 3: decoder  y[m, o] = sum_r h[m, r] * W_dec[o, r]
// One workgroup per 16-row M-tile of (B*T); 16 waves x 2 N-tiles = 512 cols.
// ---------------------------------------------------------------------------
__global__ __launch_bounds__(NTHREADS) void gru_decoder(
    const float* __restrict__ h_seq,          // (B*T, 512) f32
    const unsigned short* __restrict__ Bdec,  // swizzled bf16 W_dec
    float* __restrict__ y)                    // (B*T, 512) f32
{
  __shared__ __align__(32) unsigned short lds_a[KTILES * 32 * 16]; // 16 KB

  const int tid   = threadIdx.x;
  const int lane  = tid & 31;
  const int wave  = tid >> 5;
  const int mBase = blockIdx.x * 16;

  for (int i = tid; i < 16 * NREC; i += NTHREADS) {
    int m = i >> 9, k = i & 511;
    lds_a[a_addr(m, k)] = f2bf(h_seq[(size_t)(mBase + m) * NREC + k]);
  }
  __syncthreads();

  const v8f vzero = {0.f, 0.f, 0.f, 0.f, 0.f, 0.f, 0.f, 0.f};
  v8f acc0[NTW], acc1[NTW];
#pragma unroll
  for (int nt = 0; nt < NTW; ++nt) { acc0[nt] = vzero; acc1[nt] = vzero; }

#pragma unroll 4
  for (int kt = 0; kt < KTILES; kt += 2) {
    v16bf a0 = *(const v16bf*)&lds_a[(((kt + 0) << 5) + lane) * 16];
    v16bf a1 = *(const v16bf*)&lds_a[(((kt + 1) << 5) + lane) * 16];
#pragma unroll
    for (int nt = 0; nt < NTW; ++nt) {
      int ntile = wave * NTW + nt;
      v16bf b0 = *(const v16bf*)&Bdec[(((ntile << 4) + kt + 0) << 9) + (lane << 4)];
      acc0[nt] = wmma_bf16(a0, b0, acc0[nt]);
      v16bf b1 = *(const v16bf*)&Bdec[(((ntile << 4) + kt + 1) << 9) + (lane << 4)];
      acc1[nt] = wmma_bf16(a1, b1, acc1[nt]);
    }
  }

  const int nl = lane & 15, hiL = lane >> 4;
#pragma unroll
  for (int nt = 0; nt < NTW; ++nt) {
    int n = (wave * NTW + nt) * 16 + nl;
#pragma unroll
    for (int i = 0; i < 8; ++i) {
      int m = i + 8 * hiL;
      y[(size_t)(mBase + m) * NREC + n] = acc0[nt][i] + acc1[nt][i];
    }
  }
}

// ---------------------------------------------------------------------------
extern "C" void kernel_launch(void* const* d_in, const int* in_sizes, int n_in,
                              void* d_out, int out_size, void* d_ws, size_t ws_size,
                              hipStream_t stream) {
  const float* x          = (const float*)d_in[0];
  const float* init_state = (const float*)d_in[1];
  const float* W_enc      = (const float*)d_in[2];
  const float* W_wz       = (const float*)d_in[3];
  const float* W_uz       = (const float*)d_in[4];
  const float* W_wr       = (const float*)d_in[5];
  const float* W_ur       = (const float*)d_in[6];
  const float* W_wh       = (const float*)d_in[7];
  const float* W_uh       = (const float*)d_in[8];
  const float* W_dec      = (const float*)d_in[9];

  float* out = (float*)d_out;
  const size_t H_ELEMS = (size_t)BATCH * TT * NREC;   // 33,554,432
  float* h_out = out;
  float* y_out = out + H_ELEMS;

  unsigned short* ws = (unsigned short*)d_ws;         // 5 x 512 KB bf16 weights
  const size_t MSZ = (size_t)NREC * NREC;
  unsigned short* Benc = ws;
  unsigned short* Buz  = ws + 1 * MSZ;
  unsigned short* Bur  = ws + 2 * MSZ;
  unsigned short* Buh  = ws + 3 * MSZ;
  unsigned short* Bdec = ws + 4 * MSZ;

  dim3 cblk(256), cgrd((unsigned)(MSZ / 256));
  swizzle_weight_bf16<<<cgrd, cblk, 0, stream>>>(W_enc, Benc);
  swizzle_weight_bf16<<<cgrd, cblk, 0, stream>>>(W_uz,  Buz);
  swizzle_weight_bf16<<<cgrd, cblk, 0, stream>>>(W_ur,  Bur);
  swizzle_weight_bf16<<<cgrd, cblk, 0, stream>>>(W_uh,  Buh);
  swizzle_weight_bf16<<<cgrd, cblk, 0, stream>>>(W_dec, Bdec);

  gru_recurrence<<<BATCH / BBLK, NTHREADS, 0, stream>>>(
      x, init_state, W_wz, W_wr, W_wh, Benc, Buz, Bur, Buh, h_out);

  gru_decoder<<<(BATCH * TT) / 16, NTHREADS, 0, stream>>>(h_out, Bdec, y_out);
}
